// G2C_69715909148680
// MI455X (gfx1250) — matrix operand order repaired and checked by reading, not compiled, not run on hardware
//
#include <hip/hip_runtime.h>
#include <math.h>

// ---- problem constants (match reference) ----
#define B_       512
#define N_       32
#define H_       128
#define KRANK    3
#define PSTEPS   10
#define TSTEPS   100
#define EPS_D    0.01f
#define EPS_SPD  1e-3f
#define LRx4     0.4f      // LR(0.1) * 4 (grad factor)
#define ALPHA_   5.0f
#define ALPHAB_  0.1f

typedef __attribute__((ext_vector_type(16))) _Float16 v16h;
typedef __attribute__((ext_vector_type(8)))  float    v8f;

__device__ __forceinline__ float wred16(float v) {  // sum across 16-lane half-wave
    v += __shfl_xor(v, 1, 32);
    v += __shfl_xor(v, 2, 32);
    v += __shfl_xor(v, 4, 32);
    v += __shfl_xor(v, 8, 32);
    return v;
}
__device__ __forceinline__ float wred32(float v) {  // sum across full wave32
    v = wred16(v);
    v += __shfl_xor(v, 16, 32);
    return v;
}
__device__ __forceinline__ float softplusf(float x) {
    return (x > 20.f) ? x : log1pf(expf(x));
}

// =====================================================================
// Kernel A: fused MLP.  raw[m][0..1] = relu(e[m] @ W1 + b1) @ W2
// M = B*N*N = 524288 rows, K = 128, hidden = 128.
// Each wave: one 16-row tile; 8 nTiles x 4 kTiles of v_wmma_f32_16x16x32_f16.
// =====================================================================
__global__ __launch_bounds__(256) void mlp_wmma_kernel(
    const float* __restrict__ e,   // [M,128]
    const float* __restrict__ W1,  // [128,128] (k-major: W1[k*128+n])
    const float* __restrict__ b1,  // [128]
    const float* __restrict__ W2,  // [128,2]
    float* __restrict__ raw)       // [M,2]
{
    // W1 staged in LDS pre-swizzled into WMMA B-operand order:
    // ldsB[kT][nT][lane][el], el 0..15; K=kT*32+koff+(el&7)+(el>=8?16:0), koff=0 (lanes 0-15) / 8 (lanes 16-31)
    __shared__ __align__(32) _Float16 ldsB[4 * 8 * 32 * 16];  // 32 KB

    const int tid = threadIdx.x;
    for (int q = tid; q < 4 * 8 * 32 * 16; q += 256) {
        int el  = q & 15;
        int ln  = (q >> 4) & 31;
        int nT  = (q >> 9) & 7;
        int kT  = q >> 12;
        int kk  = kT * 32 + ((ln & 16) ? 8 : 0) + (el & 7) + ((el & 8) ? 16 : 0);
        int nn  = nT * 16 + (ln & 15);
        ldsB[q] = (_Float16)W1[kk * H_ + nn];
    }
    __syncthreads();

    const int  lane    = tid & 31;
    const int  wave    = tid >> 5;
    const long rowBase = (long)blockIdx.x * 128 + (long)wave * 16;
    const long myRow   = rowBase + (lane & 15);
    const float* rowp  = e + myRow * H_;
    const int  koff    = (lane & 16) ? 8 : 0;   // A-operand K split per half-wave

    // prefetch next tile of streamed A rows (speculative)
    if (blockIdx.x + 1 < gridDim.x)
        __builtin_prefetch(rowp + 128L * H_, 0, 1);

    // Load A tiles (16x32 f16 per kT).  Per-lane elements:
    //   el 0..7  -> K = kT*32 + koff + el
    //   el 8..15 -> K = kT*32 + koff + 16 + (el-8)
    v16h a[4];
#pragma unroll
    for (int kT = 0; kT < 4; ++kT) {
        const float4* q0 = (const float4*)(rowp + kT * 32 + koff);
        const float4* q1 = (const float4*)(rowp + kT * 32 + koff + 16);
        float4 A0 = q0[0], A1 = q0[1];
        float4 B0 = q1[0], B1 = q1[1];
        float f[16] = { A0.x, A0.y, A0.z, A0.w, A1.x, A1.y, A1.z, A1.w,
                        B0.x, B0.y, B0.z, B0.w, B1.x, B1.y, B1.z, B1.w };
#pragma unroll
        for (int t = 0; t < 16; ++t) a[kT][t] = (_Float16)f[t];
    }

    float p0[8], p1[8];
#pragma unroll
    for (int r = 0; r < 8; ++r) { p0[r] = 0.f; p1[r] = 0.f; }

#pragma unroll
    for (int nT = 0; nT < 8; ++nT) {
        v8f acc = {0.f, 0.f, 0.f, 0.f, 0.f, 0.f, 0.f, 0.f};
#pragma unroll
        for (int kT = 0; kT < 4; ++kT) {
            const v16h* bp = (const v16h*)&ldsB[(((kT * 8 + nT) * 32) + lane) * 16];
            acc = __builtin_amdgcn_wmma_f32_16x16x32_f16(
                false, a[kT], false, *bp, (short)0, acc, false, false);
        }
        // acc element r = h[row rowBase+r(+8)][col nT*16+lane&15] (pre-bias)
        const int   col = nT * 16 + (lane & 15);
        const float b1v = b1[col];
        const float w20 = W2[col * 2 + 0];
        const float w21 = W2[col * 2 + 1];
#pragma unroll
        for (int r = 0; r < 8; ++r) {
            float h = acc[r] + b1v;
            h = h > 0.f ? h : 0.f;
            p0[r] = fmaf(h, w20, p0[r]);
            p1[r] = fmaf(h, w21, p1[r]);
        }
    }

    // reduce over N (16 lanes of each half-wave hold distinct columns, same row)
#pragma unroll
    for (int r = 0; r < 8; ++r) { p0[r] = wred16(p0[r]); p1[r] = wred16(p1[r]); }

    if ((lane & 15) == 0) {
        const long mb = rowBase + ((lane >> 4) << 3);   // lane0 -> rows 0..7, lane16 -> 8..15
#pragma unroll
        for (int r = 0; r < 8; ++r) {
            raw[(mb + r) * 2 + 0] = p0[r];
            raw[(mb + r) * 2 + 1] = p1[r];
        }
    }
}

// =====================================================================
// Kernel B: per-batch solve.  1024 threads = 32 waves; wave w <-> atom i,
// lane j <-> neighbor j.  Phases: symmetrize+softplus -> D,W ; Gram ;
// rank-3 power iteration w/ deflation ; 100-step tanh-trust-region GD.
// =====================================================================
__global__ __launch_bounds__(1024) void solve_kernel(
    const float* __restrict__ raw,     // [B*N*N, 2]
    const float* __restrict__ b2,      // [2]
    const float* __restrict__ noise,   // [B,N,3]
    const float* __restrict__ u_init,  // [K,B,N,1]
    float* __restrict__ out)           // D (B*N*N) | W (B*N*N) | X (B*N*3)
{
    __shared__ float Dm[N_][N_], Wm[N_][N_], Al[N_][N_];
    __shared__ float Xs[N_][3];
    __shared__ float uu[N_], nu[N_];
    __shared__ float sbr;

    const int b   = blockIdx.x;
    const int tid = threadIdx.x;
    const int i   = tid >> 5;    // wave id = atom
    const int j   = tid & 31;    // lane    = neighbor

    // ---- phase 1: symmetrize + 2*b2 + softplus + off-diag mask ----
    {
        const long ij = (long)b * 1024 + i * 32 + j;
        const long ji = (long)b * 1024 + j * 32 + i;
        float r0 = raw[ij * 2 + 0] + raw[ji * 2 + 0] + 2.f * b2[0];
        float r1 = raw[ij * 2 + 1] + raw[ji * 2 + 1] + 2.f * b2[1];
        float Dij = softplusf(r0) * ((i == j) ? 0.f : 1.f);
        float Wij = softplusf(r1);
        Dm[i][j] = Dij;
        Wm[i][j] = Wij;
        out[ij] = Dij;
        out[(long)B_ * N_ * N_ + ij] = Wij;
    }
    __syncthreads();

    // ---- phase 2: Gram  Al = D^T D ----
    {
        float g = 0.f;
#pragma unroll
        for (int k = 0; k < N_; ++k) g = fmaf(Dm[k][i], Dm[k][j], g);
        Al[i][j] = g;
    }
    __syncthreads();

    // ---- phase 3: rank-3 power iteration with deflation ----
    for (int kk = 0; kk < KRANK; ++kk) {
        if (tid < 32) uu[tid] = u_init[((long)kk * B_ + b) * N_ + tid];
        __syncthreads();
        for (int p = 0; p < PSTEPS; ++p) {
            if (tid < 32) {
                float v = uu[tid];
                float s = wred32(v * v);
                if (tid == 0) sbr = 1.f / fmaxf(sqrtf(s), 1e-3f);
            }
            __syncthreads();
            float pr = Al[i][j] * (uu[j] * sbr);
            pr = wred32(pr);
            if (j == 0) nu[i] = pr;
            __syncthreads();
            if (tid < 32) uu[tid] = nu[tid];
            __syncthreads();
        }
        if (tid < 32) {
            float v = uu[tid];
            float s = wred32(v * v);
            if (tid == 0) sbr = powf(s + 0.01f, 0.25f);
        }
        __syncthreads();
        if (tid < 32) uu[tid] *= sbr;
        __syncthreads();
        Al[i][j] -= uu[i] * uu[j];                               // deflate
        if (tid < 32)
            Xs[tid][kk] = uu[tid] + noise[((long)b * N_ + tid) * 3 + kk];
        __syncthreads();
    }

    // ---- phase 4: 100 gradient-descent steps with tanh trust region ----
    for (int t = 0; t < TSTEPS; ++t) {
        float dx = Xs[i][0] - Xs[j][0];
        float dy = Xs[i][1] - Xs[j][1];
        float dz = Xs[i][2] - Xs[j][2];
        float dsq = fmaf(dx, dx, fmaf(dy, dy, dz * dz)) + EPS_D;
        float dij = sqrtf(dsq);
        float c   = Wm[i][j] * (Dm[i][j] - dij) / dij;
        float gx = wred32(c * dx);
        float gy = wred32(c * dy);
        float gz = wred32(c * dz);
        __syncthreads();                 // all reads of Xs complete before update
        if (j == 0) {
            float sx = LRx4 * gx, sy = LRx4 * gy, sz = LRx4 * gz;
            float speed = sqrtf(fmaf(sx, sx, fmaf(sy, sy, sz * sz)) + EPS_SPD);
            float a_t = ALPHAB_ + (ALPHA_ - ALPHAB_) * ((float)(TSTEPS - t) / (float)TSTEPS);
            float sc = a_t * tanhf(speed / a_t) / speed;
            Xs[i][0] = fmaf(sx, sc, Xs[i][0]);
            Xs[i][1] = fmaf(sy, sc, Xs[i][1]);
            Xs[i][2] = fmaf(sz, sc, Xs[i][2]);
        }
        __syncthreads();
    }

    if (tid < 32) {
        const long base = 2L * B_ * N_ * N_ + ((long)b * N_ + tid) * 3;
        out[base + 0] = Xs[tid][0];
        out[base + 1] = Xs[tid][1];
        out[base + 2] = Xs[tid][2];
    }
}

extern "C" void kernel_launch(void* const* d_in, const int* in_sizes, int n_in,
                              void* d_out, int out_size, void* d_ws, size_t ws_size,
                              hipStream_t stream) {
    (void)in_sizes; (void)n_in; (void)out_size; (void)ws_size;
    const float* e      = (const float*)d_in[0];  // [B,N,N,H]
    const float* W1     = (const float*)d_in[1];  // [H,H]
    const float* b1     = (const float*)d_in[2];  // [H]
    const float* W2     = (const float*)d_in[3];  // [H,2]
    const float* b2     = (const float*)d_in[4];  // [2]
    const float* noise  = (const float*)d_in[5];  // [B,N,3]
    const float* u_init = (const float*)d_in[6];  // [K,B,N,1]
    float*       out    = (float*)d_out;
    float*       raw    = (float*)d_ws;           // [B*N*N, 2] = 4 MB scratch

    // M = 524288 rows; 128 rows per 256-thread block -> 4096 blocks
    mlp_wmma_kernel<<<4096, 256, 0, stream>>>(e, W1, b1, W2, raw);
    solve_kernel<<<B_, 1024, 0, stream>>>(raw, b2, noise, u_init, out);
}
